// NUSpectralConv1D_13932873908828
// MI455X (gfx1250) — matrix operand order repaired and checked by reading, not compiled, not run on hardware
//
#include <hip/hip_runtime.h>
#include <math.h>

typedef _Float16 v16h __attribute__((ext_vector_type(16)));
typedef float    v8f  __attribute__((ext_vector_type(8)));

#define B_    8
#define N_    16384
#define C_    64
#define TEMB_ 256
#define M_    129
#define MPAD  160
#define H_    128

// workspace layout (float units)
constexpr size_t OFF_YR    = 0;        // _Float16[B*C*MPAD] = 81920 halfs = 40960 floats
constexpr size_t OFF_YIN   = 40960;    // _Float16[B*C*MPAD]
constexpr size_t OFF_WA    = 81920;    // float[B*N]
constexpr size_t OFF_TOFF  = 212992;   // float[B*C]
constexpr size_t OFF_G     = 213504;   // float[B*MPAD]
constexpr size_t OFF_KAPPA = 214784;   // float[MPAD]
constexpr size_t OFF_CPAD  = 214944;   // float[MPAD]
constexpr size_t OFF_OMPAD = 215104;   // float[MPAD]
constexpr size_t OFF_SCAL  = 215264;   // [q1,q2,gl,gm,gh,alpha,sptau1]
constexpr size_t OFF_DENOM = 215280;   // float[B]

__device__ __forceinline__ float softplus_f(float x) {
  return x > 20.f ? x : log1pf(__expf(x));
}
__device__ __forceinline__ float sigmoid_f(float x) {
  return 1.f / (1.f + __expf(-x));
}

__global__ void k_zero(float* p, int n) {
  int i = blockIdx.x * 256 + threadIdx.x;
  if (i < n) p[i] = 0.f;
}

// single-thread scalar precompute: kappa, quantile thresholds, gates, gabor params
__global__ void k_setup(const float* __restrict__ kraw,
                        const float* __restrict__ gab_c,
                        const float* __restrict__ gab_om,
                        const float* __restrict__ glr, const float* __restrict__ gmr,
                        const float* __restrict__ ghr,
                        const float* __restrict__ tau1r, const float* __restrict__ alphar,
                        float* __restrict__ KAPPA, float* __restrict__ CPAD,
                        float* __restrict__ OMPAD, float* __restrict__ SCAL) {
  if (threadIdx.x != 0) return;
  float kpos[64];
  float run = 0.f;
  for (int i = 0; i < 64; ++i) { run += softplus_f(kraw[i]); kpos[i] = run; }
  for (int m = 0; m < MPAD; ++m) {
    int mm = m < M_ ? m : M_ - 1;
    float kap;
    if (mm < 64)       kap = -kpos[63 - mm];
    else if (mm == 64) kap = 0.f;
    else               kap = kpos[mm - 65];
    KAPPA[m] = kap;
    CPAD[m] = gab_c[mm];
    float om = softplus_f(gab_om[mm]);
    OMPAD[m] = om < 1e-6f ? 1e-6f : om;
  }
  // |kappa| sorted = [0, kpos0,kpos0, kpos1,kpos1, ...]; linear-interp quantiles
  SCAL[0] = kpos[25];                                   // q=0.4 -> pos 51.2
  SCAL[1] = kpos[50] + 0.4f * (kpos[51] - kpos[50]);    // q=0.8 -> pos 102.4
  SCAL[2] = sigmoid_f(glr[0]);
  SCAL[3] = sigmoid_f(gmr[0]);
  SCAL[4] = sigmoid_f(ghr[0]);
  SCAL[5] = softplus_f(alphar[0]) + 1e-12f;
  SCAL[6] = softplus_f(tau1r[0]);
}

// per-batch: channel offsets from silu(temb)@tproj^T, gate MLP -> g[b,m]
__global__ __launch_bounds__(256) void k_temb(
    const float* __restrict__ temb, const float* __restrict__ tproj_w,
    const float* __restrict__ tproj_b, const float* __restrict__ mlp_w1,
    const float* __restrict__ mlp_b1, const float* __restrict__ mlp_w2,
    const float* __restrict__ mlp_b2, const float* __restrict__ tau0,
    const float* __restrict__ KAPPA, const float* __restrict__ SCAL,
    float* __restrict__ TOFF, float* __restrict__ G) {
  const int b = blockIdx.x, t = threadIdx.x;
  __shared__ float st[TEMB_];
  __shared__ float h1[H_];
  __shared__ float taus;
  float v = temb[b * TEMB_ + t];
  st[t] = v * sigmoid_f(v);   // silu
  __syncthreads();
  if (t < C_) {
    float acc = tproj_b[t];
    for (int i = 0; i < TEMB_; ++i) acc += st[i] * tproj_w[t * TEMB_ + i];
    TOFF[b * C_ + t] = acc;
  } else if (t < C_ + H_) {
    int hh = t - C_;
    float acc = mlp_b1[hh];
    for (int i = 0; i < TEMB_; ++i) acc += temb[b * TEMB_ + i] * mlp_w1[hh * TEMB_ + i];
    h1[hh] = acc * sigmoid_f(acc);
  }
  __syncthreads();
  if (t == 0) {
    float acc = mlp_b2[0];
    for (int i = 0; i < H_; ++i) acc += h1[i] * mlp_w2[i];
    taus = tau0[0] + SCAL[6] * tanhf(acc);
  }
  __syncthreads();
  if (t < MPAD) {
    float g = 0.f;
    if (t < M_) {
      float ka = fabsf(KAPPA[t]);
      float bg = ka <= SCAL[0] ? SCAL[2] : (ka <= SCAL[1] ? SCAL[3] : SCAL[4]);
      g = bg * sigmoid_f(SCAL[5] * (taus - ka));
    }
    G[b * MPAD + t] = g;
  }
}

// trapezoid quadrature: wa = sqrt(w+eps), denom = sum(w)+eps
__global__ __launch_bounds__(256) void k_quad(const float* __restrict__ z,
                                              float* __restrict__ WA,
                                              float* __restrict__ DENOM) {
  const int b = blockIdx.x, t = threadIdx.x;
  const float* zb = z + (size_t)b * N_;
  __shared__ float red[256];
  float lsum = 0.f;
  for (int n = t; n < N_; n += 256) {
    float w;
    if (n == 0)            w = 0.5f * fabsf(zb[1] - zb[0]);
    else if (n == N_ - 1)  w = 0.5f * fabsf(zb[N_ - 1] - zb[N_ - 2]);
    else                   w = 0.5f * (fabsf(zb[n + 1] - zb[n]) + fabsf(zb[n] - zb[n - 1]));
    w = fmaxf(w, 0.f);
    lsum += w;
    WA[(size_t)b * N_ + n] = sqrtf(w + 1e-12f);
  }
  red[t] = lsum;
  __syncthreads();
  for (int s = 128; s > 0; s >>= 1) {
    if (t < s) red[t] += red[t + s];
    __syncthreads();
  }
  if (t == 0) DENOM[b] = red[0] + 1e-12f;
}

// Analysis GEMM + fused gating/spectral mix.
// grid = (9 m-tiles, B), block = 256 (8 waves). Each wave owns N/8 of the
// K=N reduction with v_wmma_f32_16x16x32_f16; LDS tree combines waves.
__global__ __launch_bounds__(256) void k_analysis(
    const float* __restrict__ x_feat, const float* __restrict__ z,
    const float* __restrict__ wr, const float* __restrict__ wi,
    const float* __restrict__ WA, const float* __restrict__ TOFF,
    const float* __restrict__ G, const float* __restrict__ KAPPA,
    const float* __restrict__ CPAD, const float* __restrict__ OMPAD,
    const float* __restrict__ DENOM,
    _Float16* __restrict__ YR, _Float16* __restrict__ YIN) {
  __shared__ float red[4][2048];
  const int b = blockIdx.y, mt = blockIdx.x;
  const int lane = threadIdx.x & 31, wave = threadIdx.x >> 5;
  const int col = lane & 15, hi = lane >> 4;
  const int m = mt * 16 + col;          // B-matrix column held by this lane
  const float kap = KAPPA[m], cm = CPAD[m];
  const float inv_om = 1.f / OMPAD[m];
  const float INVPI = 0.3183098861837907f;

  const v8f z8 = {0.f, 0.f, 0.f, 0.f, 0.f, 0.f, 0.f, 0.f};
  v8f accFr[4] = {z8, z8, z8, z8};
  v8f accFi[4] = {z8, z8, z8, z8};

  const int r = col;          // A-matrix row (channel within c-tile) for this lane
  const int kh = hi * 8;      // K-half offset for A layout
  const float* zb = z + (size_t)b * N_;
  const float* wab = WA + (size_t)b * N_;
  float toffv[4];
#pragma unroll
  for (int ct = 0; ct < 4; ++ct) toffv[ct] = TOFF[b * C_ + ct * 16 + r];

  for (int chunk = wave; chunk < N_ / 32; chunk += 8) {
    const int n0 = chunk * 32;
    // B fragments: basis[n, m] computed in-register (B layout: K = (j&7) + 16*(j>>3) + 8*hi)
    v16h bc, bs;
    const int nb = n0 + hi * 8;
#pragma unroll
    for (int j = 0; j < 16; ++j) {
      float zv = zb[nb + (j & 7) + ((j >> 3) << 4)];
      float tt = (zv * INVPI - cm) * inv_om;
      float s = __expf(-0.5f * tt * tt);
      float ph = zv * kap;
      bc[j] = (_Float16)(__cosf(ph) * s);
      bs[j] = (_Float16)(__sinf(ph) * s);
    }
    float wav[16];
#pragma unroll
    for (int j = 0; j < 16; ++j) {
      int k = (j & 7) + ((j >> 3) << 4) + kh;   // A layout K permutation
      wav[j] = wab[n0 + k];
    }
#pragma unroll
    for (int ct = 0; ct < 4; ++ct) {
      const float* xp = x_feat + ((size_t)(b * C_ + ct * 16 + r)) * N_ + n0;
      v16h a;
#pragma unroll
      for (int j = 0; j < 16; ++j) {
        int k = (j & 7) + ((j >> 3) << 4) + kh;
        a[j] = (_Float16)((xp[k] + toffv[ct]) * wav[j]);
      }
      accFr[ct] = __builtin_amdgcn_wmma_f32_16x16x32_f16(
          false, a, false, bc, (short)0, accFr[ct], false, false);
      accFi[ct] = __builtin_amdgcn_wmma_f32_16x16x32_f16(
          false, a, false, bs, (short)0, accFi[ct], false, false);
    }
  }

  // deterministic LDS tree reduction across the 8 waves
  auto storeSlot = [&](int slot) {
#pragma unroll
    for (int ct = 0; ct < 4; ++ct)
#pragma unroll
      for (int rr = 0; rr < 8; ++rr) {
        red[slot][(ct * 8 + rr) * 32 + lane] = accFr[ct][rr];
        red[slot][(32 + ct * 8 + rr) * 32 + lane] = accFi[ct][rr];
      }
  };
  auto addSlot = [&](int slot) {
#pragma unroll
    for (int ct = 0; ct < 4; ++ct)
#pragma unroll
      for (int rr = 0; rr < 8; ++rr) {
        accFr[ct][rr] += red[slot][(ct * 8 + rr) * 32 + lane];
        accFi[ct][rr] += red[slot][(32 + ct * 8 + rr) * 32 + lane];
      }
  };
  if (wave >= 4) storeSlot(wave - 4);
  __syncthreads();
  if (wave < 4) addSlot(wave);
  __syncthreads();
  if (wave >= 2 && wave < 4) storeSlot(wave - 2);
  __syncthreads();
  if (wave < 2) addSlot(wave);
  __syncthreads();
  if (wave == 1) storeSlot(0);
  __syncthreads();
  if (wave == 0) {
    addSlot(0);
    const float dinv = 1.f / DENOM[b];
    const float gv = G[b * MPAD + m];
#pragma unroll
    for (int ct = 0; ct < 4; ++ct)
#pragma unroll
      for (int rr = 0; rr < 8; ++rr) {
        int c = ct * 16 + rr + 8 * hi;   // D layout: row = reg + 8*(lane>=16)
        float Fr = accFr[ct][rr] * dinv * gv;
        float Fi = -accFi[ct][rr] * dinv * gv;
        float yr = 0.f, yin = 0.f;
        if (m < M_) {
          float wrv = wr[c * M_ + m], wiv = wi[c * M_ + m];
          yr = Fr * wrv - Fi * wiv;
          yin = -(Fr * wiv + Fi * wrv);   // store -Yi for fused synthesis FMA
        }
        size_t o = ((size_t)(b * C_ + c)) * MPAD + m;
        YR[o] = (_Float16)yr;
        YIN[o] = (_Float16)yin;
      }
  }
}

// Synthesis GEMM: out[c,n] = (Yr . cos + (-Yi) . sin) * wa + bias
// grid = (N/128, B), block = 256 (8 waves, each owns 16 n-columns x 64 channels)
__global__ __launch_bounds__(256) void k_synth(
    const float* __restrict__ z, const float* __restrict__ bias,
    const float* __restrict__ WA, const float* __restrict__ KAPPA,
    const float* __restrict__ CPAD, const float* __restrict__ OMPAD,
    const _Float16* __restrict__ YR, const _Float16* __restrict__ YIN,
    float* __restrict__ out) {
  const int b = blockIdx.y;
  const int lane = threadIdx.x & 31, wave = threadIdx.x >> 5;
  const int col = lane & 15, hi = lane >> 4;
  const int n = blockIdx.x * 128 + wave * 16 + col;
  const float INVPI = 0.3183098861837907f;
  const float zv = z[(size_t)b * N_ + n];
  const float zs = zv * INVPI;

  const v8f z8 = {0.f, 0.f, 0.f, 0.f, 0.f, 0.f, 0.f, 0.f};
  v8f acc[4] = {z8, z8, z8, z8};

#pragma unroll
  for (int chunk = 0; chunk < 5; ++chunk) {   // K = MPAD = 160 (pad rows are 0)
    v16h bc, bs;
    const int mbase = chunk * 32 + hi * 8;    // B layout: K = (j&7) + 16*(j>>3) + 8*hi
#pragma unroll
    for (int j = 0; j < 16; ++j) {
      int mm = mbase + (j & 7) + ((j >> 3) << 4);
      float tt = (zs - CPAD[mm]) / OMPAD[mm];
      float s = __expf(-0.5f * tt * tt);
      float ph = zv * KAPPA[mm];
      bc[j] = (_Float16)(__cosf(ph) * s);
      bs[j] = (_Float16)(__sinf(ph) * s);
    }
#pragma unroll
    for (int ct = 0; ct < 4; ++ct) {
      const int c = ct * 16 + col;            // A row for this lane
      const size_t base = ((size_t)(b * C_ + c)) * MPAD + chunk * 32 + hi * 8;
      v16h ar, ai;
#pragma unroll
      for (int j = 0; j < 8; ++j) { ar[j] = YR[base + j];      ai[j] = YIN[base + j]; }
#pragma unroll
      for (int j = 0; j < 8; ++j) { ar[8 + j] = YR[base + 16 + j]; ai[8 + j] = YIN[base + 16 + j]; }
      acc[ct] = __builtin_amdgcn_wmma_f32_16x16x32_f16(
          false, ar, false, bc, (short)0, acc[ct], false, false);
      acc[ct] = __builtin_amdgcn_wmma_f32_16x16x32_f16(
          false, ai, false, bs, (short)0, acc[ct], false, false);
    }
  }
  const float wav = WA[(size_t)b * N_ + n];
#pragma unroll
  for (int ct = 0; ct < 4; ++ct)
#pragma unroll
    for (int rr = 0; rr < 8; ++rr) {
      int c = ct * 16 + rr + 8 * hi;          // D layout row
      out[((size_t)(b * C_ + c)) * N_ + n] = acc[ct][rr] * wav + bias[c];
    }
}

extern "C" void kernel_launch(void* const* d_in, const int* in_sizes, int n_in,
                              void* d_out, int out_size, void* d_ws, size_t ws_size,
                              hipStream_t stream) {
  (void)in_sizes; (void)n_in; (void)out_size; (void)ws_size;
  const float* x_feat  = (const float*)d_in[0];
  const float* z       = (const float*)d_in[1];
  const float* temb    = (const float*)d_in[2];
  const float* kraw    = (const float*)d_in[3];
  const float* wr      = (const float*)d_in[4];
  const float* wi      = (const float*)d_in[5];
  const float* bias    = (const float*)d_in[6];
  const float* tproj_w = (const float*)d_in[7];
  const float* tproj_b = (const float*)d_in[8];
  const float* gab_c   = (const float*)d_in[9];
  const float* gab_om  = (const float*)d_in[10];
  const float* glr     = (const float*)d_in[11];
  const float* gmr     = (const float*)d_in[12];
  const float* ghr     = (const float*)d_in[13];
  const float* mlp_w1  = (const float*)d_in[14];
  const float* mlp_b1  = (const float*)d_in[15];
  const float* mlp_w2  = (const float*)d_in[16];
  const float* mlp_b2  = (const float*)d_in[17];
  const float* tau0    = (const float*)d_in[18];
  const float* tau1r   = (const float*)d_in[19];
  const float* alphar  = (const float*)d_in[20];

  float* ws = (float*)d_ws;
  _Float16* YR  = (_Float16*)(ws + OFF_YR);
  _Float16* YIN = (_Float16*)(ws + OFF_YIN);
  float* WA    = ws + OFF_WA;
  float* TOFF  = ws + OFF_TOFF;
  float* Gbuf  = ws + OFF_G;
  float* KAPPA = ws + OFF_KAPPA;
  float* CPAD  = ws + OFF_CPAD;
  float* OMPAD = ws + OFF_OMPAD;
  float* SCAL  = ws + OFF_SCAL;
  float* DENOM = ws + OFF_DENOM;

  // zero Yr / -Yi (incl. pad m in [129,160)) so padded K-chunks contribute 0
  k_zero<<<(81920 + 255) / 256, 256, 0, stream>>>(ws + OFF_YR, 81920);
  k_setup<<<1, 32, 0, stream>>>(kraw, gab_c, gab_om, glr, gmr, ghr, tau1r, alphar,
                                KAPPA, CPAD, OMPAD, SCAL);
  k_temb<<<B_, 256, 0, stream>>>(temb, tproj_w, tproj_b, mlp_w1, mlp_b1,
                                 mlp_w2, mlp_b2, tau0, KAPPA, SCAL, TOFF, Gbuf);
  k_quad<<<B_, 256, 0, stream>>>(z, WA, DENOM);
  dim3 g4(9, B_);
  k_analysis<<<g4, 256, 0, stream>>>(x_feat, z, wr, wi, WA, TOFF, Gbuf,
                                     KAPPA, CPAD, OMPAD, DENOM, YR, YIN);
  dim3 g5(N_ / 128, B_);
  k_synth<<<g5, 256, 0, stream>>>(z, bias, WA, KAPPA, CPAD, OMPAD, YR, YIN,
                                  (float*)d_out);
}
